// MultiHeadAttention_15341623181979
// MI455X (gfx1250) — matrix operand order repaired and checked by reading, not compiled, not run on hardware
//
#include <hip/hip_runtime.h>

#define DEV __device__ __forceinline__

typedef __attribute__((ext_vector_type(16))) __bf16 v16bf;
typedef __attribute__((ext_vector_type(8)))  float  v8f;
typedef __attribute__((ext_vector_type(4)))  unsigned int u32x4;
typedef __attribute__((ext_vector_type(8)))  int i32x8;
typedef __attribute__((ext_vector_type(4)))  int i32x4;

union Frag {
  v16bf v;
  unsigned int u[8];
  uint4 q[2];
};

DEV v8f wmma_bf16(const Frag& a, const Frag& b, v8f c) {
  return __builtin_amdgcn_wmma_f32_16x16x32_bf16(false, a.v, false, b.v,
                                                 (short)0, c, false, false);
}

// 16x32 (MxK) bf16 A-fragment. base -> element (0,0); rowStride in elements.
// ISA layout: lane L (r=L%16, hl=L/16): VGPR i holds K = 8*hl + {2i,2i+1}
// for i<4 and K = 16 + 8*hl + {2(i-4), 2(i-4)+1} for i>=4.
DEV void load_frag_a(const __bf16* base, int rowStride, Frag& f) {
  int lane = (int)(threadIdx.x & 31u);
  int r = lane & 15, hl = lane >> 4;
  const __bf16* p = base + r * rowStride + 8 * hl;
#pragma unroll
  for (int i = 0; i < 8; ++i) {
    int off = (i < 4) ? (2 * i) : (16 + 2 * (i - 4));
    f.u[i] = *reinterpret_cast<const unsigned int*>(p + off);
  }
}

// 32x16 (KxN) bf16 B-fragment from N-major memory: element (k,n) at
// base[n*colStride + k]. Lane L (c=L%16, hl=L/16) holds column n=c,
// K = 8*hl + 0..7 and 16 + 8*hl + 0..7 (two b128 loads).
DEV void load_frag_b(const __bf16* base, int colStride, Frag& f) {
  int lane = (int)(threadIdx.x & 31u);
  int c = lane & 15, hl = lane >> 4;
  const __bf16* p = base + c * colStride + 8 * hl;
  f.q[0] = *reinterpret_cast<const uint4*>(p);
  f.q[1] = *reinterpret_cast<const uint4*>(p + 16);
}

// ---------------------------------------------------------------------------
// Kernel 1: QKV projection. A = hidden f32 [4096,1024], W = qkv f32 [1024,3072]
// Output: Q,K as bf16 [B,NH,S,D]; V transposed bf16 [B,NH,D,S].
// Block tile 128(M) x 128(N), K-step 32; 8 waves (4x2), each 32x64.
// ---------------------------------------------------------------------------
__global__ __launch_bounds__(256) void qkv_gemm_kernel(
    const float* __restrict__ A, const float* __restrict__ W,
    __bf16* __restrict__ Qd, __bf16* __restrict__ Kd, __bf16* __restrict__ Vt) {
  constexpr int N = 3072, K = 1024;
  __shared__ __bf16 As[128 * 32];   // row-major 128x32
  __shared__ __bf16 Bs[128 * 32];   // N-major: Bs[n][k]
  const int m0 = blockIdx.y * 128;
  const int n0 = blockIdx.x * 128;
  const int t = (int)threadIdx.x;
  const int w = t >> 5, wy = w & 3, wx = w >> 2;
  v8f acc[2][4] = {};

  for (int k0 = 0; k0 < K; k0 += 32) {
    // Stage A tile (f32 -> bf16): 16 contiguous k per thread.
    {
      int ml = t >> 1, kl = (t & 1) * 16;
      const float4* src =
          reinterpret_cast<const float4*>(A + (m0 + ml) * K + k0 + kl);
      __bf16* dst = As + ml * 32 + kl;
#pragma unroll
      for (int i = 0; i < 4; ++i) {
        float4 x = src[i];
        dst[4 * i + 0] = (__bf16)x.x; dst[4 * i + 1] = (__bf16)x.y;
        dst[4 * i + 2] = (__bf16)x.z; dst[4 * i + 3] = (__bf16)x.w;
      }
      if (k0 + 32 < K) __builtin_prefetch(A + (m0 + ml) * K + k0 + 32 + kl, 0, 1);
    }
    // Stage B tile transposed (f32 -> bf16, N-major).
    {
      int nl = t & 127, kb = (t >> 7) * 16;
#pragma unroll
      for (int i = 0; i < 16; ++i)
        Bs[nl * 32 + kb + i] = (__bf16)W[(k0 + kb + i) * N + n0 + nl];
    }
    __syncthreads();
    Frag a0, a1, b[4];
    load_frag_a(As + (wy * 32) * 32, 32, a0);
    load_frag_a(As + (wy * 32 + 16) * 32, 32, a1);
#pragma unroll
    for (int j = 0; j < 4; ++j)
      load_frag_b(Bs + (wx * 64 + j * 16) * 32, 32, b[j]);
#pragma unroll
    for (int j = 0; j < 4; ++j) {
      acc[0][j] = wmma_bf16(a0, b[j], acc[0][j]);
      acc[1][j] = wmma_bf16(a1, b[j], acc[1][j]);
    }
    __syncthreads();
  }

  // Epilogue: scatter into Q / K / V-transposed, bf16.
  const int lane = t & 31, c = lane & 15, hl = lane >> 4;
#pragma unroll
  for (int fa = 0; fa < 2; ++fa)
#pragma unroll
    for (int fb = 0; fb < 4; ++fb)
#pragma unroll
      for (int v = 0; v < 8; ++v) {
        int m = m0 + wy * 32 + fa * 16 + hl * 8 + v;
        int n = n0 + wx * 64 + fb * 16 + c;
        __bf16 bvv = (__bf16)acc[fa][fb][v];
        int tt = n >> 10, rem = n & 1023;
        int h = rem >> 6, dc = n & 63;
        int b = m >> 11, s = m & 2047;
        int bh = b * 16 + h;
        if (tt == 0)      Qd[(bh * 2048 + s) * 64 + dc] = bvv;
        else if (tt == 1) Kd[(bh * 2048 + s) * 64 + dc] = bvv;
        else              Vt[(bh * 64 + dc) * 2048 + s] = bvv;
      }
}

// ---------------------------------------------------------------------------
// Kernel 2: flash attention. Each wave owns 16 q-rows; block = 8 waves = 128
// q-rows for one (b,head). Streams K/V in 32-position tiles, online softmax.
// ---------------------------------------------------------------------------
__global__ __launch_bounds__(256) void attn_kernel(
    const __bf16* __restrict__ Qd, const __bf16* __restrict__ Kd,
    const __bf16* __restrict__ Vt, __bf16* __restrict__ attn,
    const float* __restrict__ ssm_ptr) {
  __shared__ __bf16 Ps[8][16 * 32];  // per-wave P staging (C-layout -> A-layout)
  const int t = (int)threadIdx.x;
  const int w = t >> 5, lane = t & 31, c = lane & 15, hl = lane >> 4;
  const int blk = (int)blockIdx.x;   // 512 blocks
  const int qb = blk & 15;           // q-block of 128 rows
  const int bh = blk >> 4;           // (b*16 + h)
  const int qrow0 = qb * 128 + w * 16;
  const float scale = 0.125f;        // 1/sqrt(64)
  const float ssm = ssm_ptr[0];

  const __bf16* Qbase = Qd + (bh * 2048 + qrow0) * 64;
  const __bf16* Kbase = Kd + bh * 2048 * 64;
  const __bf16* Vbase = Vt + bh * 64 * 2048;

  Frag aq0, aq1;
  load_frag_a(Qbase + 0, 64, aq0);   // d = 0..31
  load_frag_a(Qbase + 32, 64, aq1);  // d = 32..63

  v8f o[4] = {};
  float run_max[8], run_sum[8];
#pragma unroll
  for (int v = 0; v < 8; ++v) { run_max[v] = -3.0e38f; run_sum[v] = 0.0f; }

  for (int kt = 0; kt < 2048; kt += 32) {
    // Scores: S[16 x 32] = Q(16x64) . K^T(64x32), two 16-col groups.
    Frag bk00, bk01, bk10, bk11;
    load_frag_b(Kbase + (kt + 0) * 64 + 0, 64, bk00);
    load_frag_b(Kbase + (kt + 0) * 64 + 32, 64, bk01);
    load_frag_b(Kbase + (kt + 16) * 64 + 0, 64, bk10);
    load_frag_b(Kbase + (kt + 16) * 64 + 32, 64, bk11);
    v8f z = {};
    v8f c0 = wmma_bf16(aq0, bk00, z); c0 = wmma_bf16(aq1, bk01, c0);
    v8f c1 = wmma_bf16(aq0, bk10, z); c1 = wmma_bf16(aq1, bk11, c1);

    // Online softmax across the 32 new columns (row spread over 16 lanes).
#pragma unroll
    for (int v = 0; v < 8; ++v) {
      float s0 = c0[v] * scale, s1 = c1[v] * scale;
      float mx = fmaxf(s0, s1);
      mx = fmaxf(mx, __shfl_xor(mx, 1));
      mx = fmaxf(mx, __shfl_xor(mx, 2));
      mx = fmaxf(mx, __shfl_xor(mx, 4));
      mx = fmaxf(mx, __shfl_xor(mx, 8));
      float newm = fmaxf(run_max[v], mx);
      float fac = __expf(run_max[v] - newm);
      float e0 = __expf(s0 - newm), e1 = __expf(s1 - newm);
      float ts = e0 + e1;
      ts += __shfl_xor(ts, 1);
      ts += __shfl_xor(ts, 2);
      ts += __shfl_xor(ts, 4);
      ts += __shfl_xor(ts, 8);
      run_sum[v] = run_sum[v] * fac + ts;
      run_max[v] = newm;
      o[0][v] *= fac; o[1][v] *= fac; o[2][v] *= fac; o[3][v] *= fac;
      int m = hl * 8 + v;  // q-row within tile (C-layout M index)
      Ps[w][m * 32 + c] = (__bf16)e0;
      Ps[w][m * 32 + 16 + c] = (__bf16)e1;
    }
    // Same-wave LDS write->read: DS ops are in-order; block compiler motion.
    asm volatile("s_wait_dscnt 0" ::: "memory");

    // O += P(16x32) . V(32x64): V^T layout gives contiguous B fragments.
    Frag ap;
    load_frag_a(&Ps[w][0], 32, ap);
    Frag bv0, bv1, bv2, bv3;
    load_frag_b(Vbase + (0 * 16) * 2048 + kt, 2048, bv0);
    load_frag_b(Vbase + (1 * 16) * 2048 + kt, 2048, bv1);
    load_frag_b(Vbase + (2 * 16) * 2048 + kt, 2048, bv2);
    load_frag_b(Vbase + (3 * 16) * 2048 + kt, 2048, bv3);
    o[0] = wmma_bf16(ap, bv0, o[0]);
    o[1] = wmma_bf16(ap, bv1, o[1]);
    o[2] = wmma_bf16(ap, bv2, o[2]);
    o[3] = wmma_bf16(ap, bv3, o[3]);
    asm volatile("" ::: "memory");  // keep next-iter LDS writes below ap load
  }

  // Normalize, apply scaled_softmax, write attn bf16 [B,S,H].
  const int b = bh >> 4, h = bh & 15;
#pragma unroll
  for (int v = 0; v < 8; ++v) {
    float inv = ssm / run_sum[v];
    o[0][v] *= inv; o[1][v] *= inv; o[2][v] *= inv; o[3][v] *= inv;
  }
#pragma unroll
  for (int dt = 0; dt < 4; ++dt)
#pragma unroll
    for (int v = 0; v < 8; ++v) {
      int s = qrow0 + hl * 8 + v;
      int col = h * 64 + dt * 16 + c;
      attn[(b * 2048 + s) * 1024 + col] = (__bf16)o[dt][v];
    }
}

// ---------------------------------------------------------------------------
// Kernel 3: output projection. A = attn bf16 [4096,1024] staged via the
// Tensor Data Mover (D# descriptor, TENSORcnt), W = o_w f32 [1024,1024]
// converted in staging, C = f32 d_out. Block tile 128x128.
// ---------------------------------------------------------------------------
__global__ __launch_bounds__(256) void out_gemm_kernel(
    const __bf16* __restrict__ A, const float* __restrict__ W,
    float* __restrict__ out) {
  constexpr int N = 1024, K = 1024;
  __shared__ __bf16 As[128 * 32];   // row-major 128x32 (TDM destination)
  __shared__ __bf16 Bs[128 * 32];   // N-major
  const int m0 = blockIdx.y * 128;
  const int n0 = blockIdx.x * 128;
  const int t = (int)threadIdx.x;
  const int w = t >> 5, wy = w & 3, wx = w >> 2;
  v8f acc[2][4] = {};

  // TDM D# pieces that are loop-invariant.
  const unsigned lds_addr = (unsigned)(uintptr_t)(void*)As;
  const unsigned long long gbase = (unsigned long long)(uintptr_t)A;

  for (int k0 = 0; k0 < K; k0 += 32) {
    if (w == 0) {
      // D# group 0: count=1 | lds_addr | global_addr(57b) | type=2.
      unsigned long long ga = gbase + (unsigned long long)(m0 * K + k0) * 2ull;
      u32x4 g0;
      g0[0] = 1u;                                    // count=1, no gather
      g0[1] = lds_addr;                              // LDS dest (bytes)
      g0[2] = (unsigned)(ga & 0xFFFFFFFFull);        // global_addr[31:0]
      g0[3] = (unsigned)((ga >> 32) & 0x1FFFFFFull)  // global_addr[56:32]
              | (2u << 30);                          // type=2 (image)
      // D# group 1: data_size=2B; tensor [K=1024 x M=4096]; tile 32x128;
      // row stride = 1024 elements.
      i32x8 g1;
      g1[0] = 0x10000;                // data_size=1 (2 bytes), mask=0
      g1[1] = (int)((K & 0xFFFF) << 16);       // tensor_dim0[15:0] @ bits63:48
      g1[2] = (int)((4096 & 0xFFFF) << 16);    // tensor_dim1[15:0] @ bits95:80
      g1[3] = (int)(32 << 16);                 // tile_dim0=32 @ bits127:112
      g1[4] = 128;                             // tile_dim1=128 @ bits143:128
      g1[5] = K;                               // tensor_dim0_stride[31:0]
      g1[6] = 0;
      g1[7] = 0;
      i32x4 gz = {0, 0, 0, 0};
#if defined(__clang_major__) && __clang_major__ >= 23
      i32x8 gz8 = {0, 0, 0, 0, 0, 0, 0, 0};
      __builtin_amdgcn_tensor_load_to_lds(g0, g1, gz, gz, gz8, 0);
#else
      __builtin_amdgcn_tensor_load_to_lds(g0, g1, gz, gz, 0);
#endif
    }
    // All waves: stage B tile transposed (f32 -> bf16, N-major).
    {
      int nl = t & 127, kb = (t >> 7) * 16;
#pragma unroll
      for (int i = 0; i < 16; ++i)
        Bs[nl * 32 + kb + i] = (__bf16)W[(k0 + kb + i) * N + n0 + nl];
    }
    if (w == 0) __builtin_amdgcn_s_wait_tensorcnt(0);
    __syncthreads();

    Frag a0, a1, b[4];
    load_frag_a(As + (wy * 32) * 32, 32, a0);
    load_frag_a(As + (wy * 32 + 16) * 32, 32, a1);
#pragma unroll
    for (int j = 0; j < 4; ++j)
      load_frag_b(Bs + (wx * 64 + j * 16) * 32, 32, b[j]);
#pragma unroll
    for (int j = 0; j < 4; ++j) {
      acc[0][j] = wmma_bf16(a0, b[j], acc[0][j]);
      acc[1][j] = wmma_bf16(a1, b[j], acc[1][j]);
    }
    __syncthreads();
  }

  const int lane = t & 31, c = lane & 15, hl = lane >> 4;
#pragma unroll
  for (int fa = 0; fa < 2; ++fa)
#pragma unroll
    for (int fb = 0; fb < 4; ++fb)
#pragma unroll
      for (int v = 0; v < 8; ++v) {
        int m = m0 + wy * 32 + fa * 16 + hl * 8 + v;
        int n = n0 + wx * 64 + fb * 16 + c;
        out[m * N + n] = acc[fa][fb][v];
      }
}

// ---------------------------------------------------------------------------
extern "C" void kernel_launch(void* const* d_in, const int* in_sizes, int n_in,
                              void* d_out, int out_size, void* d_ws,
                              size_t ws_size, hipStream_t stream) {
  (void)in_sizes; (void)n_in; (void)out_size; (void)ws_size;
  const float* hidden = (const float*)d_in[0];   // [2,2048,1024]
  const float* qkv_w  = (const float*)d_in[1];   // [1024,3072]
  const float* o_w    = (const float*)d_in[2];   // [1024,1024]
  const float* ssm    = (const float*)d_in[4];   // scalar

  // Workspace layout (bf16): Q [B,NH,S,D] 8MB | K 8MB | V^T [B,NH,D,S] 8MB |
  // attn [B,S,H] 8MB  -> 32MB total.
  char* ws = (char*)d_ws;
  __bf16* Qd   = (__bf16*)(ws + 0);
  __bf16* Kd   = (__bf16*)(ws + (size_t)8 * 1024 * 1024);
  __bf16* Vt   = (__bf16*)(ws + (size_t)16 * 1024 * 1024);
  __bf16* attn = (__bf16*)(ws + (size_t)24 * 1024 * 1024);
  float* out = (float*)d_out;                    // [2,2048,1024] f32

  dim3 blk(256);
  qkv_gemm_kernel<<<dim3(3072 / 128, 4096 / 128), blk, 0, stream>>>(
      hidden, qkv_w, Qd, Kd, Vt);
  attn_kernel<<<dim3(512), blk, 0, stream>>>(Qd, Kd, Vt, attn, ssm);
  out_gemm_kernel<<<dim3(1024 / 128, 4096 / 128), blk, 0, stream>>>(
      attn, o_w, out);
}
